// BloomBlock_10685878633225
// MI455X (gfx1250) — matrix operand order repaired and checked by reading, not compiled, not run on hardware
//
#include <hip/hip_runtime.h>
#include <hip/hip_bf16.h>
#include <math.h>

// ---------------------------------------------------------------------------
// BLOOM block on gfx1250 (CDNA5): bf16 WMMA (f32 accum) for all matmuls.
// B=1, S=2048, H=1024, NH=16, HD=64, LAYER_NUMBER=1 -> scale = 1/8.
// GEMMs use 4x2 register blocking (64x32 per wave): 8 WMMAs per 12 b128
// loads per K=32 step, lifting L2 intensity ~2.7x over 1-tile-per-wave.
// ---------------------------------------------------------------------------

constexpr int SEQ   = 2048;
constexpr int HID   = 1024;
constexpr int NHEAD = 16;
constexpr int HDIM  = 64;
constexpr int H3    = 3 * HID;   // 3072
constexpr int H4    = 4 * HID;   // 4096

typedef __attribute__((ext_vector_type(16))) __bf16 v16bf;
typedef __attribute__((ext_vector_type(8)))  __bf16 v8bf;
typedef __attribute__((ext_vector_type(8)))  float  v8f;

// f32 -> bf16 round-to-nearest-even, stored as u16
__device__ __forceinline__ unsigned short f2bf(float f) {
    unsigned u = __float_as_uint(f);
    u = (u + 0x7FFFu + ((u >> 16) & 1u)) >> 16;
    return (unsigned short)u;
}

// Build a 16xbf16 fragment from two 16-byte chunks (global or LDS pointer).
__device__ __forceinline__ v16bf load_frag(const unsigned short* p0,
                                           const unsigned short* p1) {
    union { v16bf v; v8bf h[2]; } u;
    u.h[0] = *(const v8bf*)p0;
    u.h[1] = *(const v8bf*)p1;
    return u.v;
}

__device__ __forceinline__ v8f wmma_bf16(v16bf a, v16bf b, v8f c) {
    return __builtin_amdgcn_wmma_f32_16x16x32_bf16(
        /*neg_a=*/false, a, /*neg_b=*/false, b,
        /*c_mod=*/(short)0, c, /*reuse_a=*/false, /*reuse_b=*/false);
}

// ------------------------------ elementwise --------------------------------

__global__ void cvt_f32_bf16(const float* __restrict__ in,
                             unsigned short* __restrict__ out, int n) {
    int i = blockIdx.x * 256 + threadIdx.x;
    if (i < n) out[i] = f2bf(in[i]);
}

// One block per row; output bf16.
__global__ void ln_kernel(const float* __restrict__ x,
                          const float* __restrict__ g,
                          const float* __restrict__ b,
                          unsigned short* __restrict__ out) {
    __shared__ float ssum[256], ssq[256];
    const int row = blockIdx.x;
    const float* xr = x + (size_t)row * HID;
    float s = 0.f, q = 0.f;
    for (int i = threadIdx.x; i < HID; i += 256) {
        float v = xr[i]; s += v; q += v * v;
    }
    ssum[threadIdx.x] = s; ssq[threadIdx.x] = q;
    __syncthreads();
    for (int st = 128; st > 0; st >>= 1) {
        if ((int)threadIdx.x < st) {
            ssum[threadIdx.x] += ssum[threadIdx.x + st];
            ssq [threadIdx.x] += ssq [threadIdx.x + st];
        }
        __syncthreads();
    }
    const float mean = ssum[0] * (1.0f / HID);
    const float var  = ssq[0] * (1.0f / HID) - mean * mean;
    const float inv  = rsqrtf(var + 1e-5f);
    for (int i = threadIdx.x; i < HID; i += 256)
        out[(size_t)row * HID + i] = f2bf((xr[i] - mean) * inv * g[i] + b[i]);
}

// fused[S, NH, 3*HD] (f32) -> q[NH,S,HD], k[NH,S,HD], vT[NH,HD,S] (bf16)
__global__ void split_qkv(const float* __restrict__ qkv,
                          unsigned short* __restrict__ qb,
                          unsigned short* __restrict__ kb,
                          unsigned short* __restrict__ vtb) {
    size_t idx = (size_t)blockIdx.x * 256 + threadIdx.x;   // < SEQ*HID
    int s   = (int)(idx >> 10);
    int rem = (int)(idx & 1023);
    int h   = rem >> 6;
    int d   = rem & 63;
    size_t base = (size_t)s * H3 + (size_t)h * (3 * HDIM);
    qb [((size_t)h * SEQ + s) * HDIM + d] = f2bf(qkv[base + d]);
    kb [((size_t)h * SEQ + s) * HDIM + d] = f2bf(qkv[base + HDIM + d]);
    vtb[((size_t)h * HDIM + d) * SEQ + s] = f2bf(qkv[base + 2 * HDIM + d]);
}

// ------------------------------ WMMA GEMM ----------------------------------
// out[m,n] = sum_k A[m,k] * W[n,k]  (+ bias, epilogue)
// Wave computes a 64x32 block (4 m-tiles x 2 n-tiles). Block = 4 waves
// tiled along N -> block covers 64 rows x 128 cols.
// EPI 0: f32 out = acc + bias
// EPI 1: f32 out = acc + bias + res[m,n]
// EPI 2: bf16 out = bloom_gelu(acc + bias)
template <int EPI>
__global__ __launch_bounds__(128)
void gemm_bf16_wmma(const unsigned short* __restrict__ A,
                    const unsigned short* __restrict__ W,
                    const float* __restrict__ bias,
                    const float* __restrict__ res,
                    float* __restrict__ outF,
                    unsigned short* __restrict__ outB,
                    int M, int N, int K) {
    const int lane = threadIdx.x & 31;
    const int wave = threadIdx.x >> 5;
    const int half = lane >> 4;
    const int ln16 = lane & 15;
    const int n0 = blockIdx.x * 128 + wave * 32;   // 2 n-tiles per wave
    const int m0 = blockIdx.y * 64;                // 4 m-tiles per wave
    if (n0 + 32 > N || m0 + 64 > M) return;

    const unsigned short* Ar[4];
    const unsigned short* Wr[2];
#pragma unroll
    for (int i = 0; i < 4; ++i) Ar[i] = A + (size_t)(m0 + i * 16 + ln16) * K;
#pragma unroll
    for (int j = 0; j < 2; ++j) Wr[j] = W + (size_t)(n0 + j * 16 + ln16) * K;

    v8f acc[4][2];
#pragma unroll
    for (int i = 0; i < 4; ++i)
#pragma unroll
        for (int j = 0; j < 2; ++j) acc[i][j] = {};

    for (int k0 = 0; k0 < K; k0 += 32) {
        v16bf a[4], b[2];
#pragma unroll
        for (int i = 0; i < 4; ++i)
            a[i] = load_frag(Ar[i] + k0 + half * 16, Ar[i] + k0 + half * 16 + 8);
#pragma unroll
        for (int j = 0; j < 2; ++j)
            b[j] = load_frag(Wr[j] + k0 + half * 16, Wr[j] + k0 + half * 16 + 8);
#pragma unroll
        for (int i = 0; i < 4; ++i)
#pragma unroll
            for (int j = 0; j < 2; ++j)
                acc[i][j] = wmma_bf16(a[i], b[j], acc[i][j]);
    }

#pragma unroll
    for (int j = 0; j < 2; ++j) {
        const int n = n0 + j * 16 + ln16;
        const float bn = bias[n];
#pragma unroll
        for (int i = 0; i < 4; ++i) {
#pragma unroll
            for (int r = 0; r < 8; ++r) {
                const int m = m0 + i * 16 + r + 8 * half;
                float v = acc[i][j][r] + bn;
                if (EPI == 1) v += res[(size_t)m * N + n];
                if (EPI == 2) {
                    float x = v;
                    v = x * 0.5f * (1.0f + tanhf(0.79788456f * x * (1.0f + 0.044715f * x * x)));
                    outB[(size_t)m * N + n] = f2bf(v);
                } else {
                    outF[(size_t)m * N + n] = v;
                }
            }
        }
    }
}

// ------------------------------ attention ----------------------------------
// block = 4 waves, grid (NHEAD, SEQ/16). Dynamic LDS:
//   sc : 16 x 2048 f32 score rows (128 KB)
//   pb : 16 x 2048 bf16 prob rows (64 KB)
__global__ __launch_bounds__(128)
void attn_kernel(const unsigned short* __restrict__ qb,
                 const unsigned short* __restrict__ kb,
                 const unsigned short* __restrict__ vtb,
                 const float* __restrict__ alibi,
                 unsigned short* __restrict__ ctx) {
    extern __shared__ char smem[];
    float*          sc = (float*)smem;                                  // 16*SEQ
    unsigned short* pb = (unsigned short*)(smem + 16 * SEQ * sizeof(float));
    __shared__ float red[16][8];

    const int h    = blockIdx.x;
    const int q0   = blockIdx.y * 16;
    const int lane = threadIdx.x & 31;
    const int wave = threadIdx.x >> 5;
    const int half = lane >> 4;
    const int ln16 = lane & 15;

    const unsigned short* qh = qb  + (size_t)h * SEQ * HDIM;
    const unsigned short* kh = kb  + (size_t)h * SEQ * HDIM;
    const unsigned short* vh = vtb + (size_t)h * HDIM * SEQ;

    // ---- phase 1: scores = (Q K^T)/8 + alibi + causal mask -> LDS ----
    const unsigned short* Arow = qh + (size_t)(q0 + ln16) * HDIM;
    for (int kt = 0; kt < 32; ++kt) {
        const int k0 = wave * 512 + kt * 16;
        const unsigned short* Brow = kh + (size_t)(k0 + ln16) * HDIM;
        v8f acc = {};
#pragma unroll
        for (int kc = 0; kc < HDIM; kc += 32) {
            v16bf a = load_frag(Arow + kc + half * 16, Arow + kc + half * 16 + 8);
            v16bf b = load_frag(Brow + kc + half * 16, Brow + kc + half * 16 + 8);
            acc = wmma_bf16(a, b, acc);
        }
        const int col = k0 + ln16;
        const float al = alibi[(size_t)h * SEQ + col];
#pragma unroll
        for (int r = 0; r < 8; ++r) {
            const int m = r + 8 * half;
            float v = acc[r] * 0.125f + al;          // 1/sqrt(HD), LAYER=1
            if (col > q0 + m) v = -1e9f;             // causal mask
            sc[m * SEQ + col] = v;
        }
    }
    __syncthreads();

    // ---- phase 2: row softmax (128 threads, 8 per row) ----
    {
        const int trow = threadIdx.x >> 3;
        const int seg  = threadIdx.x & 7;
        float* rowp = sc + trow * SEQ + seg * 256;
        float mx = -3.0e38f;
        for (int i = 0; i < 256; ++i) mx = fmaxf(mx, rowp[i]);
        red[trow][seg] = mx;
        __syncthreads();
        float rmax = red[trow][0];
        for (int j = 1; j < 8; ++j) rmax = fmaxf(rmax, red[trow][j]);
        __syncthreads();
        float sum = 0.f;
        for (int i = 0; i < 256; ++i) {
            float e = __expf(rowp[i] - rmax);
            rowp[i] = e;
            sum += e;
        }
        red[trow][seg] = sum;
        __syncthreads();
        float rsum = 0.f;
        for (int j = 0; j < 8; ++j) rsum += red[trow][j];
        const float inv = 1.0f / rsum;
        unsigned short* pbrow = pb + trow * SEQ + seg * 256;
        const int qrow = q0 + trow;
        for (int i = 0; i < 256; ++i) {
            const int col = seg * 256 + i;
            pbrow[i] = (col <= qrow) ? f2bf(rowp[i] * inv) : (unsigned short)0;
        }
    }
    __syncthreads();

    // ---- phase 3: ctx tile = P @ V  (V transposed -> contiguous B frags) ----
    {
        const int c0 = wave * 16;                    // 4 waves cover HD=64
        const unsigned short* Ap = pb + ln16 * SEQ;
        const unsigned short* Bp = vh + (size_t)(c0 + ln16) * SEQ;
        v8f acc = {};
        for (int k0 = 0; k0 < SEQ; k0 += 32) {
            v16bf a = load_frag(Ap + k0 + half * 16, Ap + k0 + half * 16 + 8);
            v16bf b = load_frag(Bp + k0 + half * 16, Bp + k0 + half * 16 + 8);
            acc = wmma_bf16(a, b, acc);
        }
#pragma unroll
        for (int r = 0; r < 8; ++r) {
            const int m = q0 + r + 8 * half;
            ctx[(size_t)m * HID + h * HDIM + c0 + ln16] = f2bf(acc[r]);
        }
    }
}

// ------------------------------ launcher -----------------------------------

extern "C" void kernel_launch(void* const* d_in, const int* in_sizes, int n_in,
                              void* d_out, int out_size, void* d_ws, size_t ws_size,
                              hipStream_t stream) {
    (void)in_sizes; (void)n_in; (void)out_size; (void)ws_size;

    const float* hidden  = (const float*)d_in[0];
    // d_in[1] = attention_mask (pure causal; applied analytically)
    const float* alibi   = (const float*)d_in[2];
    const float* ln1_g   = (const float*)d_in[3];
    const float* ln1_b   = (const float*)d_in[4];
    const float* qkv_w   = (const float*)d_in[5];
    const float* qkv_b   = (const float*)d_in[6];
    const float* dense_w = (const float*)d_in[7];
    const float* dense_b = (const float*)d_in[8];
    const float* ln2_g   = (const float*)d_in[9];
    const float* ln2_b   = (const float*)d_in[10];
    const float* fc1_w   = (const float*)d_in[11];
    const float* fc1_b   = (const float*)d_in[12];
    const float* fc2_w   = (const float*)d_in[13];
    const float* fc2_b   = (const float*)d_in[14];
    float* out = (float*)d_out;

    // workspace layout (bytes)
    char* ws = (char*)d_ws;
    size_t off = 0;
    auto alloc = [&](size_t bytes) { char* p = ws + off; off += (bytes + 255) & ~(size_t)255; return p; };
    unsigned short* wqkv   = (unsigned short*)alloc((size_t)H3 * HID * 2);
    unsigned short* wdense = (unsigned short*)alloc((size_t)HID * HID * 2);
    unsigned short* wfc1   = (unsigned short*)alloc((size_t)H4 * HID * 2);
    unsigned short* wfc2   = (unsigned short*)alloc((size_t)HID * H4 * 2);
    unsigned short* ln1o   = (unsigned short*)alloc((size_t)SEQ * HID * 2);
    float*          qkvf   = (float*)         alloc((size_t)SEQ * H3  * 4);
    unsigned short* qbuf   = (unsigned short*)alloc((size_t)SEQ * HID * 2);
    unsigned short* kbuf   = (unsigned short*)alloc((size_t)SEQ * HID * 2);
    unsigned short* vtbuf  = (unsigned short*)alloc((size_t)SEQ * HID * 2);
    unsigned short* ctx    = (unsigned short*)alloc((size_t)SEQ * HID * 2);
    float*          attnO  = (float*)         alloc((size_t)SEQ * HID * 4);
    unsigned short* ln2o   = (unsigned short*)alloc((size_t)SEQ * HID * 2);
    unsigned short* hbuf   = (unsigned short*)alloc((size_t)SEQ * H4  * 2);

    // 1) weights -> bf16
    {
        int n;
        n = H3 * HID;  cvt_f32_bf16<<<(n + 255) / 256, 256, 0, stream>>>(qkv_w,   wqkv,   n);
        n = HID * HID; cvt_f32_bf16<<<(n + 255) / 256, 256, 0, stream>>>(dense_w, wdense, n);
        n = H4 * HID;  cvt_f32_bf16<<<(n + 255) / 256, 256, 0, stream>>>(fc1_w,   wfc1,   n);
        n = HID * H4;  cvt_f32_bf16<<<(n + 255) / 256, 256, 0, stream>>>(fc2_w,   wfc2,   n);
    }

    // 2) LN1
    ln_kernel<<<SEQ, 256, 0, stream>>>(hidden, ln1_g, ln1_b, ln1o);

    // 3) QKV GEMM: [S,H] x [3H,H]^T -> f32 [S,3H]
    gemm_bf16_wmma<0><<<dim3(H3 / 128, SEQ / 64), 128, 0, stream>>>(
        ln1o, wqkv, qkv_b, nullptr, qkvf, nullptr, SEQ, H3, HID);

    // 4) split + transpose V
    split_qkv<<<(SEQ * HID) / 256, 256, 0, stream>>>(qkvf, qbuf, kbuf, vtbuf);

    // 5) fused attention (scores + softmax + PV), 192 KB dynamic LDS / WG
    {
        const size_t shmem = (size_t)16 * SEQ * sizeof(float)
                           + (size_t)16 * SEQ * sizeof(unsigned short);
        attn_kernel<<<dim3(NHEAD, SEQ / 16), 128, shmem, stream>>>(
            qbuf, kbuf, vtbuf, alibi, ctx);
    }

    // 6) dense GEMM + residual -> attn_out f32
    gemm_bf16_wmma<1><<<dim3(HID / 128, SEQ / 64), 128, 0, stream>>>(
        ctx, wdense, dense_b, hidden, attnO, nullptr, SEQ, HID, HID);

    // 7) LN2
    ln_kernel<<<SEQ, 256, 0, stream>>>(attnO, ln2_g, ln2_b, ln2o);

    // 8) fc1 GEMM + bloom-gelu -> bf16 [S,4H]
    gemm_bf16_wmma<2><<<dim3(H4 / 128, SEQ / 64), 128, 0, stream>>>(
        ln2o, wfc1, fc1_b, nullptr, nullptr, hbuf, SEQ, H4, HID);

    // 9) fc2 GEMM + residual -> d_out f32
    gemm_bf16_wmma<1><<<dim3(HID / 128, SEQ / 64), 128, 0, stream>>>(
        hbuf, wfc2, fc2_b, attnO, out, nullptr, SEQ, HID, H4);
}
